// Net_67422396612616
// MI455X (gfx1250) — 8-bit matrix operand order repaired and checked by reading, compiled, not run on hardware
//
#include <hip/hip_runtime.h>


typedef __attribute__((ext_vector_type(16))) int   v16i;
typedef __attribute__((ext_vector_type(8)))  float v8f;

#define HID 128
#define BB  8192
#define TT  2048
#define BT  64
#define MT  (BT / 16)

__device__ __forceinline__ float sigf(float x) {
  x = fminf(fmaxf(x, -30.f), 30.f);
  return __builtin_amdgcn_rcpf(1.0f + __expf(-x));
}
__device__ __forceinline__ float tanhf_(float x) {
  x = fminf(fmaxf(x, -15.f), 15.f);
  return 1.0f - 2.0f * __builtin_amdgcn_rcpf(__expf(2.0f * x) + 1.0f);
}

__device__ __forceinline__ unsigned int f32_to_e4m3(float x) {
  unsigned int s = (__float_as_uint(x) >> 24) & 0x80u;
  float ax = fabsf(x);
  if (!(ax >= 0.015625f)) {
    int m = __float2int_rn(ax * 512.0f);
    if (m > 7) m = 7;
    return s | (unsigned int)m;
  }
  if (ax > 448.0f) return s | 0x7Eu;
  unsigned int u = __float_as_uint(ax);
  unsigned int lsb = (u >> 20) & 1u;
  u += 0x0007FFFFu + lsb;
  unsigned int e = (u >> 23) & 0xFFu;
  unsigned int mant = (u >> 20) & 7u;
  return s | ((e - 120u) << 3) | mant;
}

__device__ __forceinline__ unsigned int pack4_e4m3(float4 v) {
  return f32_to_e4m3(v.x) | (f32_to_e4m3(v.y) << 8) |
         (f32_to_e4m3(v.z) << 16) | (f32_to_e4m3(v.w) << 24);
}

__device__ __forceinline__ v16i load_frag16(const unsigned int* p) {
  const uint4* q = (const uint4*)p;
  uint4 a = q[0], b = q[1], c = q[2], d = q[3];
  v16i r = { (int)a.x, (int)a.y, (int)a.z, (int)a.w,
             (int)b.x, (int)b.y, (int)b.z, (int)b.w,
             (int)c.x, (int)c.y, (int)c.z, (int)c.w,
             (int)d.x, (int)d.y, (int)d.z, (int)d.w };
  return r;
}

__global__ __launch_bounds__(256, 1)
void snn_lstm_kernel(const float* __restrict__ input,
                     const float* __restrict__ W_ih1,
                     const float* __restrict__ W_hh1,
                     const float* __restrict__ b1,
                     const float* __restrict__ W_hh2,
                     const float* __restrict__ b2,
                     const float* __restrict__ W_lin,
                     const float* __restrict__ b_lin,
                     float* __restrict__ out)
{
  __shared__ unsigned int sW1[32 * 32 * 16];
  __shared__ unsigned int sW2[32 * 32 * 16];
  __shared__ unsigned int sA1[MT * 32 * 16];
  __shared__ unsigned int sA2[MT * 32 * 16];
  __shared__ unsigned int sWL[32 * 16];
  __shared__ float        sOut[BT * 32];
  __shared__ float        sx[BT];

  const int tid = threadIdx.x;
  const int w   = tid >> 5;
  const int L   = tid & 31;
  const int hl  = L >> 4;
  const int l16 = L & 15;
  const int rowBase = blockIdx.x * BT;

#pragma unroll 1
  for (int idx = tid; idx < 512 * 32; idx += 256) {
    int n = idx >> 5, kg = idx & 31, k0 = kg << 2;
    float4 w1 = *(const float4*)(W_hh1 + n * HID + k0);
    float4 w2 = *(const float4*)(W_hh2 + n * HID + k0);
    int lane = (n & 15) + (((k0 >> 4) & 1) << 4);
    int vg   = ((k0 >> 5) << 2) | (kg & 3);
    int dst  = (((n >> 4) * 32) + lane) * 16 + vg;
    sW1[dst] = pack4_e4m3(w1);
    sW2[dst] = pack4_e4m3(w2);
  }
#pragma unroll 1
  for (int idx = tid; idx < 32 * 16; idx += 256) sWL[idx] = 0u;
#pragma unroll 1
  for (int idx = tid; idx < MT * 32 * 16; idx += 256) { sA1[idx] = 0u; sA2[idx] = 0u; }
  __syncthreads();
  if (tid < 32) {
    int k0 = tid << 2;
    float4 wl = *(const float4*)(W_lin + k0);
    int lane = ((k0 >> 4) & 1) << 4;
    int vg   = ((k0 >> 5) << 2) | (tid & 3);
    sWL[lane * 16 + vg] = pack4_e4m3(wl);
  }

  const int hcol = w * 16 + l16;
  float b1r[4], b2r[4], wi1[4];
#pragma unroll
  for (int q = 0; q < 4; ++q) {
    b1r[q] = b1[q * HID + hcol];
    b2r[q] = b2[q * HID + hcol];
    wi1[q] = W_ih1[q * HID + hcol];
  }
  const float blin = b_lin[0];

  const int kA    = hcol;
  const int aHalf = (kA >> 4) & 1;
  const int aVg   = ((kA >> 5) << 2) | ((kA >> 2) & 3);
  const int aOffBase = (16 * aHalf) * 64 + aVg * 4 + (kA & 3);
  unsigned char* a1b = (unsigned char*)sA1;
  unsigned char* a2b = (unsigned char*)sA2;

  float c1[MT][8], c2[MT][8];
#pragma unroll
  for (int m = 0; m < MT; ++m)
#pragma unroll
    for (int vv = 0; vv < 8; ++vv) { c1[m][vv] = 0.f; c2[m][vv] = 0.f; }

  __syncthreads();

#pragma unroll 1
  for (int t = 0; t < TT; ++t) {
    if (tid < BT) {
      size_t base = (size_t)(rowBase + tid) * TT;
      sx[tid] = input[base + t];
      if (t + 16 < TT) __builtin_prefetch(&input[base + t + 16], 0, 0);
    }
    __syncthreads();

    unsigned int h1p[MT][2], h2p[MT][2];

    {
      v16i bf[4];
#pragma unroll
      for (int q = 0; q < 4; ++q)
        bf[q] = load_frag16(&sW1[((w + 8 * q) * 32 + L) * 16]);
#pragma unroll
      for (int m = 0; m < MT; ++m) {
        v16i af = load_frag16(&sA1[(m * 32 + L) * 16]);
        float xr[8];
#pragma unroll
        for (int vv = 0; vv < 8; ++vv) xr[vv] = sx[m * 16 + vv + 8 * hl];
        v8f acc[4];
#pragma unroll
        for (int q = 0; q < 4; ++q) {
          v8f c0;
#pragma unroll
          for (int vv = 0; vv < 8; ++vv) c0[vv] = b1r[q] + xr[vv] * wi1[q];
          acc[q] = __builtin_amdgcn_wmma_f32_16x16x128_fp8_fp8(af, bf[q], (short)0, c0, false, false);
        }
        unsigned int p0 = 0, p1 = 0;
#pragma unroll
        for (int vv = 0; vv < 8; ++vv) {
          float ig = sigf(acc[0][vv]), fg = sigf(acc[1][vv]);
          float gg = tanhf_(acc[2][vv]), og = sigf(acc[3][vv]);
          c1[m][vv] = fg * c1[m][vv] + ig * gg;
          float h = og * tanhf_(c1[m][vv]);
          unsigned int e = f32_to_e4m3(h) << ((vv & 3) * 8);
          if (vv < 4) p0 |= e; else p1 |= e;
        }
        h1p[m][0] = p0; h1p[m][1] = p1;
      }
    }

    {
      v16i bf[4];
#pragma unroll
      for (int q = 0; q < 4; ++q)
        bf[q] = load_frag16(&sW2[((w + 8 * q) * 32 + L) * 16]);
#pragma unroll
      for (int m = 0; m < MT; ++m) {
        v16i af = load_frag16(&sA2[(m * 32 + L) * 16]);
        v8f acc[4];
#pragma unroll
        for (int q = 0; q < 4; ++q) {
          v8f c0;
#pragma unroll
          for (int vv = 0; vv < 8; ++vv) c0[vv] = b2r[q];
          acc[q] = __builtin_amdgcn_wmma_f32_16x16x128_fp8_fp8(af, bf[q], (short)0, c0, false, false);
        }
        unsigned int p0 = 0, p1 = 0;
#pragma unroll
        for (int vv = 0; vv < 8; ++vv) {
          float ig = sigf(acc[0][vv]), fg = sigf(acc[1][vv]);
          float gg = tanhf_(acc[2][vv]), og = sigf(acc[3][vv]);
          c2[m][vv] = fg * c2[m][vv] + ig * gg;
          float h = og * tanhf_(c2[m][vv]);
          unsigned int e = f32_to_e4m3(h) << ((vv & 3) * 8);
          if (vv < 4) p0 |= e; else p1 |= e;
        }
        h2p[m][0] = p0; h2p[m][1] = p1;
      }
    }
    __syncthreads();

#pragma unroll
    for (int m = 0; m < MT; ++m) {
#pragma unroll
      for (int vv = 0; vv < 8; ++vv) {
        int off = m * 2048 + (vv + 8 * hl) * 64 + aOffBase;
        a1b[off] = (unsigned char)(h1p[m][vv >> 2] >> ((vv & 3) * 8));
        a2b[off] = (unsigned char)(h2p[m][vv >> 2] >> ((vv & 3) * 8));
      }
    }
    __syncthreads();

    if (w == 0) {
      v16i bf = load_frag16(&sWL[L * 16]);
#pragma unroll
      for (int m = 0; m < MT; ++m) {
        v16i af = load_frag16(&sA2[(m * 32 + L) * 16]);
        v8f cz;
#pragma unroll
        for (int vv = 0; vv < 8; ++vv) cz[vv] = blin;
        v8f oacc = __builtin_amdgcn_wmma_f32_16x16x128_fp8_fp8(af, bf, (short)0, cz, false, false);
        if (l16 == 0) {
#pragma unroll
          for (int vv = 0; vv < 8; ++vv)
            sOut[(m * 16 + vv + 8 * hl) * 32 + (t & 31)] = oacc[vv];
        }
      }
    }

    if ((t & 31) == 31) {
      __syncthreads();
      int row = tid >> 2, seg = tid & 3;
      const float4* src = (const float4*)&sOut[row * 32 + seg * 8];
      float4* dst = (float4*)&out[(size_t)(rowBase + row) * TT + (t - 31) + seg * 8];
      dst[0] = src[0];
      dst[1] = src[1];
    }
  }
}

extern "C" void kernel_launch(void* const* d_in, const int* in_sizes, int n_in,
                              void* d_out, int out_size, void* d_ws, size_t ws_size,
                              hipStream_t stream) {
  (void)in_sizes; (void)n_in; (void)out_size; (void)d_ws; (void)ws_size;
  const float* input = (const float*)d_in[0];
  const float* W_ih1 = (const float*)d_in[1];
  const float* W_hh1 = (const float*)d_in[2];
  const float* b1    = (const float*)d_in[3];
  const float* W_hh2 = (const float*)d_in[5];
  const float* b2    = (const float*)d_in[6];
  const float* W_lin = (const float*)d_in[7];
  const float* b_lin = (const float*)d_in[8];
  float* out = (float*)d_out;

  dim3 grid(BB / BT), block(256);
  hipLaunchKernelGGL(snn_lstm_kernel, grid, block, 0, stream,
                     input, W_ih1, W_hh1, b1, W_hh2, b2, W_lin, b_lin, out);
}
